// RelativeMultiHeadAttention_82678120448799
// MI455X (gfx1250) — matrix operand order repaired and checked by reading, not compiled, not run on hardware
//
#include <hip/hip_runtime.h>
#include <hip/hip_bf16.h>

// ---------------------------------------------------------------------------
// Relative Multi-Head Attention (Transformer-XL) for MI455X / gfx1250.
// B=2, L=2048, D=512, H=8, dh=64.  ~38 GFLOP total; HBM floor ~1.7us at
// 23.3 TB/s, so everything matmul-shaped runs on v_wmma_f32_16x16x32_bf16
// (bf16 operands, f32 accumulate).  Flash-style fused attention: the 256MB
// score matrices are never materialized; workspace is 20MB bf16, so the whole
// working set (32MB f32 inputs + 20MB ws) lives in the 192MB L2.
//
// V is stored head-split TRANSPOSED [B,H,dh,L] so the attention P@V
// B-fragments are direct contiguous global loads -> the attention kernel has
// no block-wide barriers at all (only wave-private, in-order LDS for the
// P-tile D-layout -> A-fragment relayout).
// ---------------------------------------------------------------------------

typedef __attribute__((ext_vector_type(16))) __bf16 v16bf;
typedef __attribute__((ext_vector_type(8)))  float  v8f;

union VBF16 {
  v16bf  v;
  __bf16 e[16];
  uint4  q[2];
};

__device__ __forceinline__ v8f wmma_bf16(v16bf a, v16bf b, v8f c) {
  return __builtin_amdgcn_wmma_f32_16x16x32_bf16(
      /*neg_a=*/false, a, /*neg_b=*/false, b,
      /*c_mod=*/(short)0, c, /*reuse_a=*/false, /*reuse_b=*/false);
}

constexpr int B_  = 2;
constexpr int L_  = 2048;
constexpr int D_  = 512;
constexpr int H_  = 8;
constexpr int DH_ = 64;
constexpr int M_  = B_ * L_;   // 4096 rows

__device__ __forceinline__ int clampL(int j) {
  return j < 0 ? 0 : (j > L_ - 1 ? L_ - 1 : j);
}

// ---------------------------------------------------------------------------
// Kernel 1: projection GEMM.  Each wave: 16(M) x 64(N) strip, so every A
// fragment feeds 4 WMMAs and the LDS weight tile (64x32 bf16, transposed) is
// shared by all 8 waves.  Block tile: 128(M) x 64(N), K-chunks of 32.
// outMode: 0 = bf16 head-split [B,H,L,dh]; 1 = bf16 head-split transposed
// [B,H,dh,L] (for V; packed b128 stores straight from the D-layout).
// ---------------------------------------------------------------------------
__global__ __launch_bounds__(256) void gemm_proj_kernel(
    const float* __restrict__ X, const float* __restrict__ W,
    const float* __restrict__ bias, __bf16* __restrict__ out,
    int hasBias, int transposedOut)
{
  __shared__ __align__(16) __bf16 WT[64][32];   // [n][k] transposed weight tile
  const int tid  = threadIdx.x;
  const int wid  = tid >> 5;
  const int lane = tid & 31;
  const int nl   = lane & 15;
  const int hi   = lane >> 4;
  const int m0   = blockIdx.x * 128 + wid * 16;
  const int n0   = blockIdx.y * 64;
  const int row  = m0 + nl;
  const int koff = hi * 16;

  v8f acc[4] = {};
  for (int k0 = 0; k0 < D_; k0 += 32) {
    __syncthreads();
#pragma unroll
    for (int i = 0; i < 8; ++i) {
      int idx = tid + i * 256;
      int kk = idx >> 6, nn = idx & 63;          // consecutive tid -> consecutive n
      WT[nn][kk] = (__bf16)W[(k0 + kk) * D_ + n0 + nn];
    }
    __syncthreads();

    // A fragment (16x32): lane<16 holds K 0..15, lane>=16 K 16..31
    VBF16 a;
    const float4* xp0 = (const float4*)(X + row * D_ + k0 + koff);
    const float4* xp1 = (const float4*)(X + row * D_ + k0 + 8 + koff);
    float4 f0 = xp0[0], f1 = xp0[1], f2 = xp1[0], f3 = xp1[1];
    a.e[0]=(__bf16)f0.x; a.e[1]=(__bf16)f0.y; a.e[2]=(__bf16)f0.z; a.e[3]=(__bf16)f0.w;
    a.e[4]=(__bf16)f1.x; a.e[5]=(__bf16)f1.y; a.e[6]=(__bf16)f1.z; a.e[7]=(__bf16)f1.w;
    a.e[8]=(__bf16)f2.x; a.e[9]=(__bf16)f2.y; a.e[10]=(__bf16)f2.z; a.e[11]=(__bf16)f2.w;
    a.e[12]=(__bf16)f3.x; a.e[13]=(__bf16)f3.y; a.e[14]=(__bf16)f3.z; a.e[15]=(__bf16)f3.w;

#pragma unroll
    for (int nt = 0; nt < 4; ++nt) {
      VBF16 b;
      const uint4* bp = (const uint4*)&WT[nt * 16 + nl][hi * 16];
      b.q[0] = bp[0]; b.q[1] = bp[1];
      acc[nt] = wmma_bf16(a.v, b.v, acc[nt]);
    }
  }

  const int bb  = m0 >> 11;
  const int ll0 = m0 & (L_ - 1);
  const int hh  = n0 >> 6;                       // n0 is a multiple of 64

  if (transposedOut) {
    // D-layout lane holds 8 consecutive M values for fixed column -> one
    // packed b128 store per n-tile gives the [B,H,dh,L] layout directly.
#pragma unroll
    for (int nt = 0; nt < 4; ++nt) {
      union { uint4 q; __bf16 e[8]; } pk;
      const float bv = hasBias ? bias[n0 + nt * 16 + nl] : 0.f;
#pragma unroll
      for (int r = 0; r < 8; ++r) pk.e[r] = (__bf16)(acc[nt][r] + bv);
      const int d = nt * 16 + nl;
      *(uint4*)(out + (((size_t)bb * H_ + hh) * DH_ + d) * L_ + ll0 + hi * 8) = pk.q;
    }
  } else {
#pragma unroll
    for (int nt = 0; nt < 4; ++nt) {
      const int dd = nt * 16 + nl;
      const float bv = hasBias ? bias[n0 + dd] : 0.f;
#pragma unroll
      for (int r = 0; r < 8; ++r) {
        int mloc = r + hi * 8;
        out[(((size_t)bb * H_ + hh) * L_ + (ll0 + mloc)) * DH_ + dd] =
            (__bf16)(acc[nt][r] + bv);
      }
    }
  }
}

// ---------------------------------------------------------------------------
// Kernel 2: fused relative attention (flash-style, exact relative shift).
// Grid: (L/128, B*H); each wave owns a 16-row q tile, streams 32-col chunks.
// shifted[q,c] = U[q, c-q+L-1] for c<=q ; 0 for c==q+1 ; U[q+1, c-q-2] else,
// U = Q @ P^T.  The j-windows of a chunk's two 16-col tiles overlap by 16, so
// one 48-wide strip (3 j-tiles) per strip-kind covers the whole chunk; the
// diagonal band is gathered cross-lane with __shfl (ds_bpermute).
// WMMA/chunk: 4 content + 12 strip + 4 PV = 20.  No block barriers.
// ---------------------------------------------------------------------------
__global__ __launch_bounds__(256) void attn_kernel(
    const __bf16* __restrict__ QH, const __bf16* __restrict__ KH,
    const __bf16* __restrict__ VT, const __bf16* __restrict__ PH,
    __bf16* __restrict__ CTX)
{
  __shared__ __align__(16) __bf16 PL[8][16][32];   // per-wave P relayout (8KB)

  const int tid  = threadIdx.x;
  const int wid  = tid >> 5;
  const int lane = tid & 31;
  const int nl   = lane & 15;
  const int hi   = lane >> 4;
  const int bh   = blockIdx.y;
  const int q0   = blockIdx.x * 128 + wid * 16;

  const __bf16* qp = QH + (size_t)bh * L_ * DH_;
  const __bf16* kp = KH + (size_t)bh * L_ * DH_;
  const __bf16* vp = VT + (size_t)bh * DH_ * L_;   // transposed [dh][L]
  const __bf16* pp = PH + (size_t)bh * L_ * DH_;

  const int arow  = q0 + nl;
  const int arow1 = (arow + 1 < L_) ? (arow + 1) : (L_ - 1);  // wrap row (clamp unused)
  const int koff  = hi * 16;

  // Loop-invariant A fragments over dh=64 (two K-chunks of 32)
  VBF16 aq[2], aqs[2];
#pragma unroll
  for (int kc = 0; kc < 2; ++kc) {
    aq[kc].q[0]  = *(const uint4*)(qp + arow  * DH_ + kc * 32 + koff);
    aq[kc].q[1]  = *(const uint4*)(qp + arow  * DH_ + kc * 32 + 8 + koff);
    aqs[kc].q[0] = *(const uint4*)(qp + arow1 * DH_ + kc * 32 + koff);
    aqs[kc].q[1] = *(const uint4*)(qp + arow1 * DH_ + kc * 32 + 8 + koff);
  }

  v8f o0 = {}, o1 = {}, o2 = {}, o3 = {};     // ctx accumulators 16 x 64
  float mrow[8], srow[8];
#pragma unroll
  for (int r = 0; r < 8; ++r) { mrow[r] = -1e30f; srow[r] = 0.f; }

  const float scale = 0.044194173824159216f;  // 1/sqrt(512)

  for (int cc = 0; cc < L_; cc += 32) {
    // --- pos strips: 48-wide j window shared by both 16-col tiles ---
    v8f UA0 = {}, UA1 = {}, UA2 = {}, UB0 = {}, UB1 = {}, UB2 = {};
    const int jA = cc - q0 + L_ - 16;   // rows q
    const int jB = cc - q0 - 17;        // rows q+1 (wrap)
#pragma unroll
    for (int kc = 0; kc < 2; ++kc) {
      VBF16 b;
      int j;
      j = clampL(jA + nl);
      { const uint4* bp = (const uint4*)(pp + j * DH_ + kc * 32 + hi * 16); b.q[0]=bp[0]; b.q[1]=bp[1]; }
      UA0 = wmma_bf16(aq[kc].v, b.v, UA0);
      j = clampL(jA + 16 + nl);
      { const uint4* bp = (const uint4*)(pp + j * DH_ + kc * 32 + hi * 16); b.q[0]=bp[0]; b.q[1]=bp[1]; }
      UA1 = wmma_bf16(aq[kc].v, b.v, UA1);
      j = clampL(jA + 32 + nl);
      { const uint4* bp = (const uint4*)(pp + j * DH_ + kc * 32 + hi * 16); b.q[0]=bp[0]; b.q[1]=bp[1]; }
      UA2 = wmma_bf16(aq[kc].v, b.v, UA2);
      j = clampL(jB + nl);
      { const uint4* bp = (const uint4*)(pp + j * DH_ + kc * 32 + hi * 16); b.q[0]=bp[0]; b.q[1]=bp[1]; }
      UB0 = wmma_bf16(aqs[kc].v, b.v, UB0);
      j = clampL(jB + 16 + nl);
      { const uint4* bp = (const uint4*)(pp + j * DH_ + kc * 32 + hi * 16); b.q[0]=bp[0]; b.q[1]=bp[1]; }
      UB1 = wmma_bf16(aqs[kc].v, b.v, UB1);
      j = clampL(jB + 32 + nl);
      { const uint4* bp = (const uint4*)(pp + j * DH_ + kc * 32 + hi * 16); b.q[0]=bp[0]; b.q[1]=bp[1]; }
      UB2 = wmma_bf16(aqs[kc].v, b.v, UB2);
    }

    float lg[2][8];
#pragma unroll
    for (int t = 0; t < 2; ++t) {
      const int c0t = cc + t * 16;

      // content = q . k^T   (2 WMMAs over dh)
      v8f s = {};
#pragma unroll
      for (int kc = 0; kc < 2; ++kc) {
        VBF16 b;
        const uint4* bp = (const uint4*)(kp + (c0t + nl) * DH_ + kc * 32 + hi * 16);
        b.q[0] = bp[0]; b.q[1] = bp[1];
        s = wmma_bf16(aq[kc].v, b.v, s);
      }

      // gather shifted pos from the strips (cross-lane diagonal band)
#pragma unroll
      for (int r = 0; r < 8; ++r) {
        const int mloc = r + hi * 8;
        const int m    = q0 + mloc;
        const int c    = c0t + nl;
        const int col  = t * 16 + nl + 15 - mloc;   // 0..46 within 48-wide strip
        const int src  = (col & 15) + (hi << 4);    // holder lane (same half)
        float a0 = __shfl(UA0[r], src, 32);
        float a1 = __shfl(UA1[r], src, 32);
        float a2 = __shfl(UA2[r], src, 32);
        float b0 = __shfl(UB0[r], src, 32);
        float b1 = __shfl(UB1[r], src, 32);
        float b2 = __shfl(UB2[r], src, 32);
        float uA = (col < 16) ? a0 : ((col < 32) ? a1 : a2);
        float uB = (col < 16) ? b0 : ((col < 32) ? b1 : b2);
        float posv = (c <= m) ? uA : ((c == m + 1) ? 0.f : uB);
        lg[t][r] = (s[r] + posv) * scale;
      }
    }

    // online softmax update + stage P (bf16) in wave-private LDS
    float alpha[8];
#pragma unroll
    for (int r = 0; r < 8; ++r) {
      float tm = fmaxf(lg[0][r], lg[1][r]);
#pragma unroll
      for (int msk = 1; msk < 16; msk <<= 1) tm = fmaxf(tm, __shfl_xor(tm, msk, 32));
      float mnew = fmaxf(mrow[r], tm);
      alpha[r] = __expf(mrow[r] - mnew);
      float p0 = __expf(lg[0][r] - mnew);
      float p1 = __expf(lg[1][r] - mnew);
      float rs = p0 + p1;
#pragma unroll
      for (int msk = 1; msk < 16; msk <<= 1) rs += __shfl_xor(rs, msk, 32);
      srow[r] = srow[r] * alpha[r] + rs;
      mrow[r] = mnew;
      const int mloc = r + hi * 8;
      PL[wid][mloc][nl]      = (__bf16)p0;
      PL[wid][mloc][16 + nl] = (__bf16)p1;
      o0[r] *= alpha[r]; o1[r] *= alpha[r]; o2[r] *= alpha[r]; o3[r] *= alpha[r];
    }

    // P @ V : B-fragments straight from transposed V in global (contiguous 32B)
    VBF16 pa;
    pa.q[0] = *(const uint4*)&PL[wid][nl][koff];
    pa.q[1] = *(const uint4*)&PL[wid][nl][8 + koff];
#pragma unroll
    for (int nt = 0; nt < 4; ++nt) {
      VBF16 b;
      const uint4* bp = (const uint4*)(vp + ((size_t)(nt * 16 + nl)) * L_ + cc + hi * 16);
      b.q[0] = bp[0]; b.q[1] = bp[1];
      v8f& o = (nt == 0 ? o0 : nt == 1 ? o1 : nt == 2 ? o2 : o3);
      o = wmma_bf16(pa.v, b.v, o);
    }
  }

  // epilogue: normalize by softmax sum, store bf16 head-split ctx
  __bf16* op = CTX + (size_t)bh * L_ * DH_;
#pragma unroll
  for (int nt = 0; nt < 4; ++nt) {
    const v8f& o = (nt == 0 ? o0 : nt == 1 ? o1 : nt == 2 ? o2 : o3);
#pragma unroll
    for (int r = 0; r < 8; ++r) {
      int mloc = r + hi * 8;
      int m = q0 + mloc;
      int d = nt * 16 + nl;
      op[m * DH_ + d] = (__bf16)(o[r] / srow[r]);
    }
  }
}

// ---------------------------------------------------------------------------
// Kernel 3: output GEMM.  out[m,n] (f32) = ctx[m,:] @ Wo[:,n] + bo[n],
// ctx read from bf16 head-split layout.  Same 16x64-per-wave tiling.
// ---------------------------------------------------------------------------
__global__ __launch_bounds__(256) void gemm_out_kernel(
    const __bf16* __restrict__ CTXW, const float* __restrict__ W,
    const float* __restrict__ bias, float* __restrict__ out)
{
  __shared__ __align__(16) __bf16 WT[64][32];
  const int tid  = threadIdx.x;
  const int wid  = tid >> 5;
  const int lane = tid & 31;
  const int nl   = lane & 15;
  const int hi   = lane >> 4;
  const int m0   = blockIdx.x * 128 + wid * 16;
  const int n0   = blockIdx.y * 64;
  const int row  = m0 + nl;
  const int koff = hi * 16;
  const int bb   = row >> 11;
  const int ll   = row & (L_ - 1);

  v8f acc[4] = {};
  for (int k0 = 0; k0 < D_; k0 += 32) {
    __syncthreads();
#pragma unroll
    for (int i = 0; i < 8; ++i) {
      int idx = tid + i * 256;
      int kk = idx >> 6, nn = idx & 63;
      WT[nn][kk] = (__bf16)W[(k0 + kk) * D_ + n0 + nn];
    }
    __syncthreads();

    VBF16 a;
    const int k1 = k0 + koff;
    const int k2 = k0 + 8 + koff;
    a.q[0] = *(const uint4*)(CTXW + (((size_t)bb * H_ + (k1 >> 6)) * L_ + ll) * DH_ + (k1 & 63));
    a.q[1] = *(const uint4*)(CTXW + (((size_t)bb * H_ + (k2 >> 6)) * L_ + ll) * DH_ + (k2 & 63));

#pragma unroll
    for (int nt = 0; nt < 4; ++nt) {
      VBF16 b;
      const uint4* bp = (const uint4*)&WT[nt * 16 + nl][hi * 16];
      b.q[0] = bp[0]; b.q[1] = bp[1];
      acc[nt] = wmma_bf16(a.v, b.v, acc[nt]);
    }
  }

#pragma unroll
  for (int nt = 0; nt < 4; ++nt) {
    const int n = n0 + nt * 16 + nl;
    const float bv = bias[n];
#pragma unroll
    for (int r = 0; r < 8; ++r) {
      int m = m0 + r + hi * 8;
      out[(size_t)m * D_ + n] = acc[nt][r] + bv;
    }
  }
}

// ---------------------------------------------------------------------------
extern "C" void kernel_launch(void* const* d_in, const int* in_sizes, int n_in,
                              void* d_out, int out_size, void* d_ws, size_t ws_size,
                              hipStream_t stream) {
  const float* query = (const float*)d_in[0];
  const float* key_  = (const float*)d_in[1];
  const float* value = (const float*)d_in[2];
  const float* posE  = (const float*)d_in[3];
  const float* Wq = (const float*)d_in[4];
  const float* bq = (const float*)d_in[5];
  const float* Wk = (const float*)d_in[6];
  const float* bk = (const float*)d_in[7];
  const float* Wv = (const float*)d_in[8];
  const float* bv = (const float*)d_in[9];
  const float* Wp = (const float*)d_in[10];
  const float* Wo = (const float*)d_in[11];
  const float* bo = (const float*)d_in[12];
  float* out = (float*)d_out;

  const size_t tensBytes = (size_t)B_ * H_ * L_ * DH_ * 2;  // 4 MiB each
  char* ws = (char*)d_ws;
  __bf16* qh  = (__bf16*)(ws + 0 * tensBytes);
  __bf16* kh  = (__bf16*)(ws + 1 * tensBytes);
  __bf16* vt  = (__bf16*)(ws + 2 * tensBytes);   // transposed [B,H,dh,L]
  __bf16* ph  = (__bf16*)(ws + 3 * tensBytes);
  __bf16* ctx = (__bf16*)(ws + 4 * tensBytes);

  dim3 blk(256);
  dim3 gGemm(M_ / 128, D_ / 64);   // 32 x 8
  gemm_proj_kernel<<<gGemm, blk, 0, stream>>>(query, Wq, bq, qh, 1, 0);
  gemm_proj_kernel<<<gGemm, blk, 0, stream>>>(key_,  Wk, bk, kh, 1, 0);
  gemm_proj_kernel<<<gGemm, blk, 0, stream>>>(value, Wv, bv, vt, 1, 1);  // transposed
  gemm_proj_kernel<<<gGemm, blk, 0, stream>>>(posE,  Wp, bq, ph, 0, 0);  // no bias

  dim3 gAttn(L_ / 128, B_ * H_);   // 16 x 16
  attn_kernel<<<gAttn, blk, 0, stream>>>(qh, kh, vt, ph, ctx);

  gemm_out_kernel<<<gGemm, blk, 0, stream>>>(ctx, Wo, bo, out);
}
